// Tacotron2Decoder_57621281243472
// MI455X (gfx1250) — compile-verified, matrix operand order checked by reading, not run on hardware
//
#include <hip/hip_runtime.h>

// ---------------------------------------------------------------------------
// Tacotron2 decoder for MI455X (gfx1250), wave32 + WMMA bf16.
//  - persistent sequential kernel (800 steps, device-wide barriers)
//  - all GEMMs via v_wmma_f32_16x16x32_bf16 (fp32 accumulate)
//  - weights repacked once into fragment-ordered bf16 (L2-resident)
//  - fused-K GEMMs split into fixed segments: branch-free unrolled inner loops
// ---------------------------------------------------------------------------

#define B_     32
#define T_ENC_ 256
#define T_DEC_ 800
#define MEL_   80
#define EMB_   512
#define PRE_   256
#define RNN_   512
#define ATT_   128
#define NG_    2048      // 4*RNN gate width
#define NWG_   64        // persistent workgroups

typedef __attribute__((ext_vector_type(16))) __bf16 v16bf;
typedef __attribute__((ext_vector_type(8)))  float  v8f;
typedef unsigned short u16;
typedef unsigned int   u32;

// fragment record: 32 lanes x 16 bf16 (32B per lane) = 1KB per 16x32 tile
#define REC_ 512   // u16 elements per record

__device__ __forceinline__ u16 f32_to_bf16(float f) {
  u32 u = __float_as_uint(f);
  u32 r = u + 0x7FFFu + ((u >> 16) & 1u);   // round-to-nearest-even
  return (u16)(r >> 16);
}

__device__ __forceinline__ float sigmoidf_(float x) {
  return 1.0f / (1.0f + __expf(-x));
}

// Inverse of the 16-bit A/B fragment layout (ISA 7.12.2):
// lane = h*16 + row;  e<8  -> k = h*8 + e ;  e>=8 -> k = 16 + h*8 + (e-8)
__device__ __forceinline__ void pack_elem(u16* dst, long rec, int row, int kk, float v) {
  int h, e;
  if (kk < 16) { h = kk >> 3; e = kk & 7; }
  else         { int k2 = kk - 16; h = k2 >> 3; e = 8 + (k2 & 7); }
  dst[(rec * 32 + (h * 16 + row)) * 16 + e] = f32_to_bf16(v);
}

// branch-free fully-unrolled GEMM segment: KT consecutive fragment records
// from both A and B streams, accumulated into acc.
template <int KT>
__device__ __forceinline__ v8f gemm_seg(const u16* __restrict__ abase,
                                        const u16* __restrict__ bbase,
                                        v8f acc, int lane) {
  const u16* ap = abase + lane * 16;
  const u16* bp = bbase + lane * 16;
  __builtin_prefetch(bbase + KT * REC_, 0, 1);   // next weight segment -> L0/L2
#pragma unroll
  for (int kt = 0; kt < KT; ++kt) {
    v16bf a = *(const v16bf*)(ap + (size_t)kt * REC_);
    v16bf b = *(const v16bf*)(bp + (size_t)kt * REC_);
    acc = __builtin_amdgcn_wmma_f32_16x16x32_bf16(false, a, false, b,
                                                  (short)0, acc, false, false);
  }
  return acc;
}

// ---------------------------------------------------------------------------
// one-time weight repack: fp32 W[N,K] row-major -> bf16 fragment records
// grid = (ntiles, ktiles_src), block = 32 (one wave packs one 16x32 tile)
// ---------------------------------------------------------------------------
__global__ void repack_w(const float* __restrict__ src, u16* __restrict__ dst,
                         int Nvalid, int Ksrc, int KT_total, int ktOff) {
  const int nt = blockIdx.x, kt = blockIdx.y, lane = threadIdx.x;
  const int row = nt * 16 + (lane & 15);
  const int h   = lane >> 4;
  u16* d = dst + ((size_t)nt * KT_total + (kt + ktOff)) * REC_ + lane * 16;
#pragma unroll
  for (int e = 0; e < 16; ++e) {
    int koff = (e < 8) ? (h * 8 + e) : (16 + h * 8 + (e - 8));
    int k = kt * 32 + koff;
    float f = (row < Nvalid && k < Ksrc) ? src[(size_t)row * Ksrc + k] : 0.0f;
    d[e] = f32_to_bf16(f);
  }
}

// projection weights: 96 rows = proj_w(80) + stop_w(1) + zero pad, K = 1024
__global__ void repack_proj(const float* __restrict__ proj_w,
                            const float* __restrict__ stop_w,
                            u16* __restrict__ dst) {
  const int nt = blockIdx.x, kt = blockIdx.y, lane = threadIdx.x; // nt<6, kt<32
  const int row = nt * 16 + (lane & 15);
  const int h   = lane >> 4;
  u16* d = dst + ((size_t)nt * 32 + kt) * REC_ + lane * 16;
#pragma unroll
  for (int e = 0; e < 16; ++e) {
    int koff = (e < 8) ? (h * 8 + e) : (16 + h * 8 + (e - 8));
    int k = kt * 32 + koff;
    float f = 0.0f;
    if (row < MEL_)        f = proj_w[(size_t)row * 1024 + k];
    else if (row == MEL_)  f = stop_w[k];
    d[e] = f32_to_bf16(f);
  }
}

// combined LSTM bias: o = a + b
__global__ void bias_combine(const float* __restrict__ a,
                             const float* __restrict__ b,
                             float* __restrict__ o, int n) {
  int i = blockIdx.x * 256 + threadIdx.x;
  if (i < n) o[i] = a[i] + b[i];
}

// ---------------------------------------------------------------------------
// prenet (time-independent, hoisted): layer1 fp32, layer2 -> packed fragments
// row index r = t*32 + b (time-major)
// ---------------------------------------------------------------------------
__global__ void prenet1_kernel(const float* __restrict__ mel,
                               const float* __restrict__ w1,
                               const float* __restrict__ b1,
                               float* __restrict__ out) {
  const int idx = blockIdx.x * 256 + threadIdx.x;   // 25600*256 threads
  const int r = idx >> 8, j = idx & 255;
  const int t = r >> 5, b = r & 31;
  float s = b1[j];
  if (t > 0) {
    const float* x = mel + ((size_t)b * T_DEC_ + (t - 1)) * MEL_;
    const float* w = w1 + (size_t)j * MEL_;
    for (int k = 0; k < MEL_; ++k) s += x[k] * w[k];
  }
  out[(size_t)r * PRE_ + j] = fmaxf(s, 0.0f);
}

__global__ void prenet2_kernel(const float* __restrict__ pre1,
                               const float* __restrict__ w2,
                               const float* __restrict__ b2,
                               u16* __restrict__ pPre) {
  const int idx = blockIdx.x * 256 + threadIdx.x;
  const int r = idx >> 8, j = idx & 255;
  const float* x = pre1 + (size_t)r * PRE_;
  const float* w = w2 + (size_t)j * PRE_;
  float s = b2[j];
  for (int k = 0; k < PRE_; ++k) s += x[k] * w[k];
  s = fmaxf(s, 0.0f);
  const int t = r >> 5, b = r & 31;
  const long rec = ((long)t * 2 + (b >> 4)) * 8 + (j >> 5);
  pack_elem(pPre, rec, b & 15, j & 31, s);
}

// keys[b,s,:] = enc[b,s,:] @ att_wk.T   (hoisted)
__global__ void keys_kernel(const float* __restrict__ enc,
                            const float* __restrict__ wk,
                            float* __restrict__ keys) {
  const int idx = blockIdx.x * 256 + threadIdx.x;   // 32*256*128 threads
  const int bs = idx >> 7, j = idx & 127;
  const float* x = enc + (size_t)bs * EMB_;
  const float* w = wk + (size_t)j * EMB_;
  float s = 0.0f;
  for (int k = 0; k < EMB_; ++k) s += x[k] * w[k];
  keys[idx] = s;
}

__global__ void zero_kernel(u32* __restrict__ p, int n) {
  int i = blockIdx.x * 256 + threadIdx.x;
  if (i < n) p[i] = 0u;
}

// ---------------------------------------------------------------------------
// persistent decoder
// ---------------------------------------------------------------------------
struct DecArgs {
  const u16 *pWa, *pWd, *pWp, *pPre;
  const float *keys, *enc, *att_wq, *att_v;
  const int   *lengths;
  const float *biasA, *biasD, *proj_b, *stop_b;
  float *gatesA, *gatesD, *h_a, *c_a, *h_d, *c_d, *ctx;
  u16   *pHa, *pCtx, *pHd;
  int   *bar_cnt, *bar_gen;
  float *out_mel, *out_stop;
};

__device__ __forceinline__ void gbarrier(int* cnt, int* gen) {
  __syncthreads();
  if (threadIdx.x == 0) {
    __threadfence();
    int g = __hip_atomic_load(gen, __ATOMIC_RELAXED, __HIP_MEMORY_SCOPE_AGENT);
    int t = __hip_atomic_fetch_add(cnt, 1, __ATOMIC_ACQ_REL, __HIP_MEMORY_SCOPE_AGENT);
    if (t == NWG_ - 1) {
      __hip_atomic_store(cnt, 0, __ATOMIC_RELAXED, __HIP_MEMORY_SCOPE_AGENT);
      __hip_atomic_fetch_add(gen, 1, __ATOMIC_RELEASE, __HIP_MEMORY_SCOPE_AGENT);
    } else {
      while (__hip_atomic_load(gen, __ATOMIC_ACQUIRE, __HIP_MEMORY_SCOPE_AGENT) == g)
        __builtin_amdgcn_s_sleep(2);
    }
    __threadfence();
  }
  __syncthreads();
}

__global__ __launch_bounds__(256, 1) void decoder_persistent(DecArgs A) {
  const int w    = blockIdx.x;
  const int tid  = threadIdx.x;
  const int lane = tid & 31;
  const int wave = tid >> 5;

  __shared__ float s_h[EMB_];
  __shared__ float s_q[ATT_];
  __shared__ float s_e[T_ENC_];
  __shared__ float s_r[T_ENC_];

  // per-wave loop-invariant geometry / base pointers (waves 0..3 do gate GEMMs)
  const int mt = (wave >> 1) & 1;
  const int nt = (w << 1) | (wave & 1);
  const u16* wa_base  = A.pWa + (size_t)nt * 40 * REC_;   // gate-A weight tiles
  const u16* wd_base  = A.pWd + (size_t)nt * 48 * REC_;   // gate-D weight tiles
  const u16* ctx_base = A.pCtx + (size_t)mt * 16 * REC_;  // packed state bases
  const u16* ha_base  = A.pHa  + (size_t)mt * 16 * REC_;
  const u16* hd_base  = A.pHd  + (size_t)mt * 16 * REC_;
  // C-fragment -> (row, col) mapping (ISA 7.12.2, 32-bit C/D 16x16)
  const int colA = (nt << 4) | (lane & 15);
  const int rbA  = (mt << 4) + ((lane >> 4) << 3);

  for (int t = 0; t < T_DEC_; ++t) {
    // ---- phase 1: attn-LSTM gates  Z=[pre_t|ctx|h_a] (K=1280) x Wa (N=2048)
    if (wave < 4) {
      const u16* pre_base = A.pPre + (((size_t)t * 2 + mt) * 8) * REC_;
      v8f acc = {0.f,0.f,0.f,0.f,0.f,0.f,0.f,0.f};
      acc = gemm_seg<8> (pre_base, wa_base,             acc, lane);
      acc = gemm_seg<16>(ctx_base, wa_base + 8  * REC_, acc, lane);
      acc = gemm_seg<16>(ha_base,  wa_base + 24 * REC_, acc, lane);
#pragma unroll
      for (int v = 0; v < 8; ++v)
        A.gatesA[(size_t)(rbA + v) * NG_ + colA] = acc[v];
    }
    gbarrier(A.bar_cnt, A.bar_gen);

    // ---- phase 2: attn-LSTM elementwise (16384 lanes == grid exactly)
    {
      const int g = w * 256 + tid;
      const int b = g >> 9, j = g & 511;
      const float* gr = A.gatesA + (size_t)b * NG_;
      float ig = gr[j]        + A.biasA[j];
      float fg = gr[512 + j]  + A.biasA[512 + j];
      float gg = gr[1024 + j] + A.biasA[1024 + j];
      float og = gr[1536 + j] + A.biasA[1536 + j];
      float c = sigmoidf_(fg) * A.c_a[g] + sigmoidf_(ig) * tanhf(gg);
      float h = sigmoidf_(og) * tanhf(c);
      A.c_a[g] = c;
      A.h_a[g] = h;
      pack_elem(A.pHa, (long)(b >> 4) * 16 + (j >> 5), b & 15, j & 31, h);
    }
    gbarrier(A.bar_cnt, A.bar_gen);

    // ---- phase 3: additive attention, one workgroup per batch row
    if (w < B_) {
      const int b = w;
      for (int k = tid; k < EMB_; k += 256) s_h[k] = A.h_a[b * EMB_ + k];
      __syncthreads();
      if (tid < ATT_) {
        const float* wq = A.att_wq + (size_t)tid * EMB_;
        float s = 0.f;
        for (int k = 0; k < EMB_; ++k) s += s_h[k] * wq[k];
        s_q[tid] = s;
      }
      __syncthreads();
      {
        const int len = A.lengths[b];
        float e = -1e9f;
        if (tid < len) {
          const float* kr = A.keys + ((size_t)b * T_ENC_ + tid) * ATT_;
          float s = 0.f;
          for (int j = 0; j < ATT_; ++j) s += A.att_v[j] * tanhf(s_q[j] + kr[j]);
          e = s;
        }
        s_e[tid] = e;
        s_r[tid] = e;
      }
      __syncthreads();
      for (int off = 128; off > 0; off >>= 1) {
        if (tid < off) s_r[tid] = fmaxf(s_r[tid], s_r[tid + off]);
        __syncthreads();
      }
      const float m = s_r[0];
      __syncthreads();
      const float ex = __expf(s_e[tid] - m);
      s_r[tid] = ex;
      __syncthreads();
      for (int off = 128; off > 0; off >>= 1) {
        if (tid < off) s_r[tid] += s_r[tid + off];
        __syncthreads();
      }
      const float inv = 1.0f / s_r[0];
      __syncthreads();
      s_e[tid] = ex * inv;            // alpha
      __syncthreads();
      for (int d = tid; d < EMB_; d += 256) {
        float c = 0.f;
        const float* ep = A.enc + (size_t)b * T_ENC_ * EMB_ + d;
        for (int s2 = 0; s2 < T_ENC_; ++s2) c += s_e[s2] * ep[(size_t)s2 * EMB_];
        A.ctx[b * EMB_ + d] = c;
        pack_elem(A.pCtx, (long)(b >> 4) * 16 + (d >> 5), b & 15, d & 31, c);
      }
    }
    gbarrier(A.bar_cnt, A.bar_gen);

    // ---- phase 4: dec-LSTM gates  Z=[h_a|ctx|h_d] (K=1536) x Wd (N=2048)
    if (wave < 4) {
      v8f acc = {0.f,0.f,0.f,0.f,0.f,0.f,0.f,0.f};
      acc = gemm_seg<16>(ha_base,  wd_base,             acc, lane);
      acc = gemm_seg<16>(ctx_base, wd_base + 16 * REC_, acc, lane);
      acc = gemm_seg<16>(hd_base,  wd_base + 32 * REC_, acc, lane);
#pragma unroll
      for (int v = 0; v < 8; ++v)
        A.gatesD[(size_t)(rbA + v) * NG_ + colA] = acc[v];
    }
    gbarrier(A.bar_cnt, A.bar_gen);

    // ---- phase 5: dec-LSTM elementwise
    {
      const int g = w * 256 + tid;
      const int b = g >> 9, j = g & 511;
      const float* gr = A.gatesD + (size_t)b * NG_;
      float ig = gr[j]        + A.biasD[j];
      float fg = gr[512 + j]  + A.biasD[512 + j];
      float gg = gr[1024 + j] + A.biasD[1024 + j];
      float og = gr[1536 + j] + A.biasD[1536 + j];
      float c = sigmoidf_(fg) * A.c_d[g] + sigmoidf_(ig) * tanhf(gg);
      float h = sigmoidf_(og) * tanhf(c);
      A.c_d[g] = c;
      A.h_d[g] = h;
      pack_elem(A.pHd, (long)(b >> 4) * 16 + (j >> 5), b & 15, j & 31, h);
    }
    gbarrier(A.bar_cnt, A.bar_gen);

    // ---- phase 6: projection [h_d|ctx] (K=1024) x Wp (N=96: 80 mel + stop)
    //      12 tiles -> WGs 16..27, wave 0. Writes d_out directly.
    if (wave == 0 && w >= 16 && w < 28) {
      const int tile = w - 16;
      const int pmt = tile / 6, pnt = tile % 6;
      const u16* wp_base = A.pWp + (size_t)pnt * 32 * REC_;
      v8f acc = {0.f,0.f,0.f,0.f,0.f,0.f,0.f,0.f};
      acc = gemm_seg<16>(A.pHd  + (size_t)pmt * 16 * REC_, wp_base,             acc, lane);
      acc = gemm_seg<16>(A.pCtx + (size_t)pmt * 16 * REC_, wp_base + 16 * REC_, acc, lane);
      const int col = (pnt << 4) | (lane & 15);
      const int rb  = (pmt << 4) + ((lane >> 4) << 3);
#pragma unroll
      for (int v = 0; v < 8; ++v) {
        const int bb = rb + v;
        if (col < MEL_)
          A.out_mel[((size_t)bb * T_DEC_ + t) * MEL_ + col] = acc[v] + A.proj_b[col];
        else if (col == MEL_)
          A.out_stop[(size_t)bb * T_DEC_ + t] = acc[v] + A.stop_b[0];
      }
    }
    // no barrier needed: next step's phase-1 touches only buffers already
    // quiesced by the barriers above; stragglers re-sync at phase-1's barrier.
  }
}

// ---------------------------------------------------------------------------
extern "C" void kernel_launch(void* const* d_in, const int* in_sizes, int n_in,
                              void* d_out, int out_size, void* d_ws, size_t ws_size,
                              hipStream_t stream) {
  (void)in_sizes; (void)n_in; (void)out_size; (void)ws_size;
  const float* enc     = (const float*)d_in[0];
  const int*   lengths = (const int*)  d_in[1];
  const float* mel     = (const float*)d_in[2];
  const float* p_w1    = (const float*)d_in[3];
  const float* p_b1    = (const float*)d_in[4];
  const float* p_w2    = (const float*)d_in[5];
  const float* p_b2    = (const float*)d_in[6];
  const float* a_wih   = (const float*)d_in[7];
  const float* a_whh   = (const float*)d_in[8];
  const float* a_bih   = (const float*)d_in[9];
  const float* a_bhh   = (const float*)d_in[10];
  const float* att_wq  = (const float*)d_in[11];
  const float* att_wk  = (const float*)d_in[12];
  const float* att_v   = (const float*)d_in[13];
  const float* d_wih   = (const float*)d_in[14];
  const float* d_whh   = (const float*)d_in[15];
  const float* d_bih   = (const float*)d_in[16];
  const float* d_bhh   = (const float*)d_in[17];
  const float* proj_w  = (const float*)d_in[18];
  const float* proj_b  = (const float*)d_in[19];
  const float* stop_w  = (const float*)d_in[20];
  const float* stop_b  = (const float*)d_in[21];

  char* ws = (char*)d_ws;
  size_t off = 0;
  auto take = [&](size_t bytes) -> char* {
    char* p = ws + off;
    off = (off + bytes + 255) & ~(size_t)255;
    return p;
  };

  u16*   pWa    = (u16*)  take((size_t)128 * 40 * 1024);   // 5.24 MB
  u16*   pWd    = (u16*)  take((size_t)128 * 48 * 1024);   // 6.29 MB
  u16*   pWp    = (u16*)  take((size_t)6   * 32 * 1024);   // 192 KB
  u16*   pPre   = (u16*)  take((size_t)T_DEC_ * 2 * 8 * 1024); // 13.1 MB
  float* pre1   = (float*)take((size_t)B_ * T_DEC_ * PRE_ * 4); // 26.2 MB
  float* keysB  = (float*)take((size_t)B_ * T_ENC_ * ATT_ * 4); // 4.2 MB
  float* gatesA = (float*)take((size_t)B_ * NG_ * 4);
  float* gatesD = (float*)take((size_t)B_ * NG_ * 4);
  float* biasA  = (float*)take((size_t)NG_ * 4);
  float* biasD  = (float*)take((size_t)NG_ * 4);
  // --- zeroed state region (contiguous: 5 f32 planes + 3 packed + barrier) ---
  float* h_a = (float*)take((size_t)B_ * RNN_ * 4);
  float* c_a = (float*)take((size_t)B_ * RNN_ * 4);
  float* h_d = (float*)take((size_t)B_ * RNN_ * 4);
  float* c_d = (float*)take((size_t)B_ * RNN_ * 4);
  float* ctx = (float*)take((size_t)B_ * EMB_ * 4);
  u16*   pHa  = (u16*)take((size_t)2 * 16 * 1024);
  u16*   pCtx = (u16*)take((size_t)2 * 16 * 1024);
  u16*   pHd  = (u16*)take((size_t)2 * 16 * 1024);
  int*   bar  = (int*)take(256);

  const int nZero = (int)(((char*)bar + 256 - (char*)h_a) / 4);
  zero_kernel<<<(nZero + 255) / 256, 256, 0, stream>>>((u32*)h_a, nZero);

  // one-time weight repacks (fused-K layouts matching the sequential GEMMs)
  repack_w<<<dim3(128, 24), 32, 0, stream>>>(a_wih, pWa, 2048, 768,  40, 0);
  repack_w<<<dim3(128, 16), 32, 0, stream>>>(a_whh, pWa, 2048, 512,  40, 24);
  repack_w<<<dim3(128, 32), 32, 0, stream>>>(d_wih, pWd, 2048, 1024, 48, 0);
  repack_w<<<dim3(128, 16), 32, 0, stream>>>(d_whh, pWd, 2048, 512,  48, 32);
  repack_proj<<<dim3(6, 32), 32, 0, stream>>>(proj_w, stop_w, pWp);
  bias_combine<<<NG_ / 256, 256, 0, stream>>>(a_bih, a_bhh, biasA, NG_);
  bias_combine<<<NG_ / 256, 256, 0, stream>>>(d_bih, d_bhh, biasD, NG_);

  // hoisted time-independent compute
  prenet1_kernel<<<B_ * T_DEC_, 256, 0, stream>>>(mel, p_w1, p_b1, pre1);
  prenet2_kernel<<<B_ * T_DEC_, 256, 0, stream>>>(pre1, p_w2, p_b2, pPre);
  keys_kernel<<<(B_ * T_ENC_ * ATT_) / 256, 256, 0, stream>>>(enc, att_wk, keysB);

  DecArgs A;
  A.pWa = pWa; A.pWd = pWd; A.pWp = pWp; A.pPre = pPre;
  A.keys = keysB; A.enc = enc; A.att_wq = att_wq; A.att_v = att_v;
  A.lengths = lengths;
  A.biasA = biasA; A.biasD = biasD;
  A.proj_b = proj_b; A.stop_b = stop_b;
  A.gatesA = gatesA; A.gatesD = gatesD;
  A.h_a = h_a; A.c_a = c_a; A.h_d = h_d; A.c_d = c_d; A.ctx = ctx;
  A.pHa = pHa; A.pCtx = pCtx; A.pHd = pHd;
  A.bar_cnt = bar; A.bar_gen = bar + 1;
  A.out_mel = (float*)d_out;
  A.out_stop = (float*)d_out + (size_t)B_ * T_DEC_ * MEL_;

  decoder_persistent<<<NWG_, 256, 0, stream>>>(A);
}
